// ArcStandard_86758339379654
// MI455X (gfx1250) — matrix operand order repaired and checked by reading, not compiled, not run on hardware
//
#include <hip/hip_runtime.h>
#include <hip/hip_bf16.h>

// ---------------------------------------------------------------------------
// Sizes (fixed by the reference): B=32, T=256, D=512, H=512, TD=128, NT=84, R=40
// Persistent single-WGP design: 1024 threads = 32 wave32 waves.
// GEMMs: M=32 -> 2 WMMA M-tiles; N-tiles spread over the 32 waves.
// WMMA k-loops use unroll(2): 2 MMAs + 2 load batches in flight per iteration,
// still within the ~128 VGPR/wave budget (32 waves resident on the WGP).
// Tile loops are driven by a readfirstlane'd wave index so loop control is
// scalar (SALU) and EXEC stays all-ones around WMMA as the ISA requires.
// ---------------------------------------------------------------------------

typedef __bf16 bfx;
typedef __attribute__((ext_vector_type(16))) bfx  v16bf;
typedef __attribute__((ext_vector_type(8)))  float v8f;
typedef __attribute__((ext_vector_type(4)))  unsigned int u32x4;

static __device__ __forceinline__ bfx f2bf(float f) {
  unsigned u = __builtin_bit_cast(unsigned, f);
  unsigned r = u + 0x7FFFu + ((u >> 16) & 1u);
  unsigned short h = (unsigned short)(r >> 16);
  return __builtin_bit_cast(bfx, h);
}
static __device__ __forceinline__ float bf2f(bfx v) {
  unsigned short h = __builtin_bit_cast(unsigned short, v);
  unsigned u = ((unsigned)h) << 16;
  return __builtin_bit_cast(float, u);
}
static __device__ __forceinline__ float sigm(float x) { return 1.0f / (1.0f + expf(-x)); }

struct B32x2 { u32x4 a, b; };
static __device__ __forceinline__ v16bf ld2x16B(const bfx* p0, const bfx* p1) {
  B32x2 t;
  t.a = *reinterpret_cast<const u32x4*>(p0);
  t.b = *reinterpret_cast<const u32x4*>(p1);
  return __builtin_bit_cast(v16bf, t);
}
// A-tile (16x32 bf16, MxK): lane<16 -> K=[kb..kb+7] & [kb+16..kb+23]; lane>=16 -> +8.
static __device__ __forceinline__ v16bf loadA(const bfx* row, int kb, int lh) {
  const bfx* p = row + kb + lh * 8;
  return ld2x16B(p, p + 16);
}
// B-tile (32x16 bf16, KxN) with B[k][n] = W[n][k]; wrow = W + n*K + kb (contiguous in k).
static __device__ __forceinline__ v16bf loadB(const bfx* wrow, int lh) {
  const bfx* p = wrow + lh * 8;
  return ld2x16B(p, p + 16);
}
static __device__ __forceinline__ v8f wmma_bf(v16bf a, v16bf b, v8f c) {
  return __builtin_amdgcn_wmma_f32_16x16x32_bf16(false, a, false, b, (short)0, c, false, false);
}
static __device__ __forceinline__ v8f bias8(float v) {
  v8f a;
#pragma unroll
  for (int r = 0; r < 8; r++) a[r] = v;
  return a;
}
// C/D layout: VGPR r -> row m = mt*16 + lh*8 + r, col n = nt*16 + lm. g stride = 2048.
static __device__ __forceinline__ void storeC(float* g, v8f acc, int mt, int nt, int lm, int lh) {
  int m0 = mt * 16 + lh * 8, n = nt * 16 + lm;
#pragma unroll
  for (int r = 0; r < 8; r++) g[(m0 + r) * 2048 + n] = acc[r];
}
static __device__ __forceinline__ void gsync() { __threadfence(); __syncthreads(); }

// ---------------------------------------------------------------------------
// Prep kernels (run once per launch)
// ---------------------------------------------------------------------------
__global__ void k_f2bf(const float* __restrict__ s, bfx* __restrict__ d, int n) {
  int i = blockIdx.x * blockDim.x + threadIdx.x;
  if (i < n) d[i] = f2bf(s[i]);
}
__global__ void k_padW3(const float* __restrict__ s, bfx* __restrict__ d) {
  int i = blockIdx.x * blockDim.x + threadIdx.x;
  if (i < 96 * 832) {
    int row = i / 832;
    d[i] = (row < 84) ? f2bf(s[i]) : f2bf(0.0f);
  }
}
__global__ void k_biassum(const float* __restrict__ a, const float* __restrict__ b,
                          float* __restrict__ d, int n) {
  int i = blockIdx.x * blockDim.x + threadIdx.x;
  if (i < n) d[i] = a[i] + b[i];
}
// xr[b][t][k] = x[b][clip(len[b]-1-t,0,T-1)][k]  (reversed, length-aligned gather) -> bf16
__global__ void k_gatherx(const float* __restrict__ x, const int* __restrict__ lengths,
                          bfx* __restrict__ xr) {
  int i = blockIdx.x * blockDim.x + threadIdx.x;
  if (i < 32 * 256 * 512) {
    int b = i >> 17;               // / (256*512)
    int r = i & (131072 - 1);
    int t = r >> 9, k = r & 511;
    int st = lengths[b] - 1 - t;
    st = st < 0 ? 0 : (st > 255 ? 255 : st);
    xr[i] = f2bf(x[((size_t)b * 256 + st) * 512 + k]);
  }
}
__global__ void k_init(const float* __restrict__ stk_begin, const float* __restrict__ trn_begin,
                       bfx* s1, bfx* s2, bfx* sh1h, bfx* sh2h, float* shc1, float* shc2,
                       bfx* thh, float* thc) {
  int i = blockIdx.x * blockDim.x + threadIdx.x;
  if (i < 32 * 512) {
    int j = i & 511;
    s1[i] = f2bf(0.0f);
    s2[i] = f2bf(0.0f);
    sh1h[i] = f2bf(stk_begin[j]);
    sh2h[i] = f2bf(stk_begin[j]);
    shc1[i] = stk_begin[512 + j];
    shc2[i] = stk_begin[512 + j];
  }
  if (i < 32 * 128) {
    int j = i & 127;
    thh[i] = f2bf(trn_begin[j]);
    thc[i] = trn_begin[128 + j];
  }
}

// ---------------------------------------------------------------------------
// Phase 1: buffer LSTM scan over 256 steps. 1 workgroup, 32 waves.
// Gate buffer g (32x2048 f32) lives in LDS; cell state c in L2-resident ws;
// hidden state re-read from HbH (bf16) so no per-thread arrays live across
// the WMMA region (keeps VGPR pressure low -> no scratch spills).
// ---------------------------------------------------------------------------
__global__ __launch_bounds__(1024, 1) void phase1_kernel(
    const bfx* __restrict__ xr, bfx* __restrict__ HbH, float* __restrict__ cbuf,
    const bfx* __restrict__ Wih, const bfx* __restrict__ Whh,
    const float* __restrict__ bsum, const float* __restrict__ begin,
    const int* __restrict__ lengths) {
  extern __shared__ char smem[];
  float* g = (float*)smem;                       // 32*2048 f32 = 256 KB LDS
  const int tid = threadIdx.x, lane = tid & 31;
  const int lm = lane & 15, lh = lane >> 4;
  const int wv = __builtin_amdgcn_readfirstlane(tid >> 5);  // wave-uniform (SGPR)

  for (int e = tid; e < 16384; e += 1024) {
    int b = e >> 9, j = e & 511;
    HbH[(size_t)b * 257 * 512 + j] = f2bf(begin[j]);   // Hb row 0 = h0
    cbuf[e] = begin[512 + j];
  }
  gsync();

  for (int t = 0; t < 256; t++) {
    // g = xr_t @ Wih.T + h_t @ Whh.T + (bih+bhh)   -- 256 tiles, 8 per wave
    for (int tile = wv; tile < 256; tile += 32) {
      int mt = tile & 1, nt = tile >> 1;
      int m = mt * 16 + lm, n = nt * 16 + lm;
      v8f acc = bias8(bsum[n]);
      const bfx* rx = xr + ((size_t)m * 256 + t) * 512;
      const bfx* rh = HbH + ((size_t)m * 257 + t) * 512;
      const bfx* wi = Wih + (size_t)n * 512;
      const bfx* wh = Whh + (size_t)n * 512;
#pragma unroll 2
      for (int k = 0; k < 512; k += 32) acc = wmma_bf(loadA(rx, k, lh), loadB(wi + k, lh), acc);
#pragma unroll 2
      for (int k = 0; k < 512; k += 32) acc = wmma_bf(loadA(rh, k, lh), loadB(wh + k, lh), acc);
      storeC(g, acc, mt, nt, lm, lh);
    }
    __syncthreads();
    // LSTM elementwise + sequence masking (t < lengths[b] keeps update)
    for (int e = tid; e < 16384; e += 1024) {
      int b = e >> 9, j = e & 511;
      const float* gr = g + b * 2048;
      float c_old = cbuf[e];
      float h_old = bf2f(HbH[((size_t)b * 257 + t) * 512 + j]);
      float c2 = sigm(gr[512 + j]) * c_old + sigm(gr[j]) * tanhf(gr[1024 + j]);
      float hn = sigm(gr[1536 + j]) * tanhf(c2);
      bool keep = t < lengths[b];
      float hv = keep ? hn : h_old;
      cbuf[e] = keep ? c2 : c_old;
      HbH[((size_t)b * 257 + t + 1) * 512 + j] = f2bf(hv);
    }
    gsync();
  }
}

// ---------------------------------------------------------------------------
// Phase 2: 384 parser steps. 1 workgroup, 32 waves, LDS gate buffer reused
// by every GEMM in the chain. Gathers (buf_ptr/action) resolved per A-row.
// ---------------------------------------------------------------------------
struct P2 {
  bfx *HbH, *s1, *s2, *sh1h, *sh2h, *thh, *z1, *z2, *compbf, *tsh, *trh;
  float *shc1, *shc2, *thc, *tsc, *trc;
  const bfx *W1, *W2, *W3, *compW, *stkWih, *stkWhh, *trnWih, *trnWhh, *transbf, *relbf;
  const float *b1, *b2, *b3, *compb, *stkB, *trnB, *mask;
  const int *lengths, *nsteps;
  float* out;
};

__global__ __launch_bounds__(1024, 1) void phase2_kernel(P2 p) {
  extern __shared__ char smem[];
  float* g = (float*)smem;                       // 32*2048 f32 = 256 KB LDS
  __shared__ int actL[32], shiftL[32], bptrL[32], bptr2L[32];
  const int tid = threadIdx.x, lane = tid & 31;
  const int lm = lane & 15, lh = lane >> 4;
  const int wv = __builtin_amdgcn_readfirstlane(tid >> 5);  // wave-uniform (SGPR)

  if (tid < 32) bptrL[tid] = p.lengths[tid];
  gsync();

  const int NS = p.nsteps[0];
  for (int step = 0; step < NS; step++) {
    // ---- cls1: feats(1664) = [buf_top(512) | sh1_h(512) | sh2_h(512) | th_h(128)]
    for (int tile = wv; tile < 104; tile += 32) {
      int mt = tile & 1, nt = tile >> 1;
      int m = mt * 16 + lm, n = nt * 16 + lm;
      v8f acc = bias8(p.b1[n]);
      const bfx* rA0 = p.HbH + ((size_t)m * 257 + bptrL[m]) * 512;  // buf_top gather
      const bfx* rA1 = p.sh1h + m * 512;
      const bfx* rA2 = p.sh2h + m * 512;
      const bfx* rA3 = p.thh + m * 128;
      const bfx* wr = p.W1 + (size_t)n * 1664;
#pragma unroll 2
      for (int k = 0; k < 512; k += 32) acc = wmma_bf(loadA(rA0, k, lh), loadB(wr + k, lh), acc);
#pragma unroll 2
      for (int k = 0; k < 512; k += 32) acc = wmma_bf(loadA(rA1, k, lh), loadB(wr + 512 + k, lh), acc);
#pragma unroll 2
      for (int k = 0; k < 512; k += 32) acc = wmma_bf(loadA(rA2, k, lh), loadB(wr + 1024 + k, lh), acc);
#pragma unroll 2
      for (int k = 0; k < 128; k += 32) acc = wmma_bf(loadA(rA3, k, lh), loadB(wr + 1536 + k, lh), acc);
      storeC(g, acc, mt, nt, lm, lh);
    }
    __syncthreads();
    for (int e = tid; e < 32 * 832; e += 1024) {
      int b = e / 832, j = e - b * 832;
      p.z1[e] = f2bf(sigm(g[b * 2048 + j]));
    }
    gsync();
    // ---- cls2
    for (int tile = wv; tile < 104; tile += 32) {
      int mt = tile & 1, nt = tile >> 1;
      int m = mt * 16 + lm, n = nt * 16 + lm;
      v8f acc = bias8(p.b2[n]);
      const bfx* rA = p.z1 + m * 832;
      const bfx* wr = p.W2 + (size_t)n * 832;
#pragma unroll 2
      for (int k = 0; k < 832; k += 32) acc = wmma_bf(loadA(rA, k, lh), loadB(wr + k, lh), acc);
      storeC(g, acc, mt, nt, lm, lh);
    }
    __syncthreads();
    for (int e = tid; e < 32 * 832; e += 1024) {
      int b = e / 832, j = e - b * 832;
      p.z2[e] = f2bf(sigm(g[b * 2048 + j]));
    }
    gsync();
    // ---- cls3 (N padded 84 -> 96 with zero rows)
    for (int tile = wv; tile < 12; tile += 32) {
      int mt = tile & 1, nt = tile >> 1;
      int m = mt * 16 + lm, n = nt * 16 + lm;
      v8f acc = bias8(n < 84 ? p.b3[n] : 0.0f);
      const bfx* rA = p.z2 + m * 832;
      const bfx* wr = p.W3 + (size_t)n * 832;
#pragma unroll 2
      for (int k = 0; k < 832; k += 32) acc = wmma_bf(loadA(rA, k, lh), loadB(wr + k, lh), acc);
      storeC(g, acc, mt, nt, lm, lh);
    }
    __syncthreads();
    // ---- log_softmax + argmax + logp output; one thread per batch row
    if (tid < 32) {
      int b = tid;
      const float* gr = g + b * 2048;
      const float* mk = p.mask + b * 84;
      float mx = -3.4e38f;
      for (int n2 = 0; n2 < 84; n2++) { float v = gr[n2] * mk[n2]; mx = fmaxf(mx, v); }
      float s = 0.0f;
      for (int n2 = 0; n2 < 84; n2++) { float v = gr[n2] * mk[n2]; s += expf(v - mx); }
      float lse = logf(s);
      float best = -3.4e38f; int bi = 0;
      float* ob = p.out + ((size_t)step * 32 + b) * 84;
      for (int n2 = 0; n2 < 84; n2++) {
        float v = gr[n2] * mk[n2];
        float lp = v - mx - lse;
        ob[n2] = lp;
        if (lp > best) { best = lp; bi = n2; }
      }
      actL[b] = bi;
      int sh = (bi == 0 && bptrL[b] > 0) ? 1 : 0;
      shiftL[b] = sh;
      bptr2L[b] = sh ? bptrL[b] - 1 : bptrL[b];
    }
    gsync();
    // ---- comp = tanh([s1|s2|rel_emb[action%40]] @ comp_W.T + comp_b)
    for (int tile = wv; tile < 64; tile += 32) {
      int mt = tile & 1, nt = tile >> 1;
      int m = mt * 16 + lm, n = nt * 16 + lm;
      v8f acc = bias8(p.compb[n]);
      const bfx* rs1 = p.s1 + m * 512;
      const bfx* rs2 = p.s2 + m * 512;
      const bfx* rrl = p.relbf + (size_t)(actL[m] % 40) * 256;
      const bfx* wr = p.compW + (size_t)n * 1280;
#pragma unroll 2
      for (int k = 0; k < 512; k += 32) acc = wmma_bf(loadA(rs1, k, lh), loadB(wr + k, lh), acc);
#pragma unroll 2
      for (int k = 0; k < 512; k += 32) acc = wmma_bf(loadA(rs2, k, lh), loadB(wr + 512 + k, lh), acc);
#pragma unroll 2
      for (int k = 0; k < 256; k += 32) acc = wmma_bf(loadA(rrl, k, lh), loadB(wr + 1024 + k, lh), acc);
      storeC(g, acc, mt, nt, lm, lh);
    }
    __syncthreads();
    for (int e = tid; e < 16384; e += 1024)
      p.compbf[e] = f2bf(tanhf(g[(e >> 9) * 2048 + (e & 511)]));
    gsync();
    // ---- stacked LSTM, shift branch: sh1_s = LSTM(buf_top, sh1)
    for (int tile = wv; tile < 256; tile += 32) {
      int mt = tile & 1, nt = tile >> 1;
      int m = mt * 16 + lm, n = nt * 16 + lm;
      v8f acc = bias8(p.stkB[n]);
      const bfx* rx = p.HbH + ((size_t)m * 257 + bptrL[m]) * 512;
      const bfx* rh = p.sh1h + m * 512;
      const bfx* wi = p.stkWih + (size_t)n * 512;
      const bfx* wh = p.stkWhh + (size_t)n * 512;
#pragma unroll 2
      for (int k = 0; k < 512; k += 32) acc = wmma_bf(loadA(rx, k, lh), loadB(wi + k, lh), acc);
#pragma unroll 2
      for (int k = 0; k < 512; k += 32) acc = wmma_bf(loadA(rh, k, lh), loadB(wh + k, lh), acc);
      storeC(g, acc, mt, nt, lm, lh);
    }
    __syncthreads();
    for (int e = tid; e < 16384; e += 1024) {
      int b = e >> 9, j = e & 511;
      const float* gr = g + b * 2048;
      float c2 = sigm(gr[512 + j]) * p.shc1[e] + sigm(gr[j]) * tanhf(gr[1024 + j]);
      p.tsh[e] = f2bf(sigm(gr[1536 + j]) * tanhf(c2));
      p.tsc[e] = c2;
    }
    gsync();
    // ---- stacked LSTM, reduce branch: sh1_r = LSTM(comp, sh2)
    for (int tile = wv; tile < 256; tile += 32) {
      int mt = tile & 1, nt = tile >> 1;
      int m = mt * 16 + lm, n = nt * 16 + lm;
      v8f acc = bias8(p.stkB[n]);
      const bfx* rx = p.compbf + m * 512;
      const bfx* rh = p.sh2h + m * 512;
      const bfx* wi = p.stkWih + (size_t)n * 512;
      const bfx* wh = p.stkWhh + (size_t)n * 512;
#pragma unroll 2
      for (int k = 0; k < 512; k += 32) acc = wmma_bf(loadA(rx, k, lh), loadB(wi + k, lh), acc);
#pragma unroll 2
      for (int k = 0; k < 512; k += 32) acc = wmma_bf(loadA(rh, k, lh), loadB(wh + k, lh), acc);
      storeC(g, acc, mt, nt, lm, lh);
    }
    __syncthreads();
    for (int e = tid; e < 16384; e += 1024) {
      int b = e >> 9, j = e & 511;
      const float* gr = g + b * 2048;
      float c2 = sigm(gr[512 + j]) * p.shc2[e] + sigm(gr[j]) * tanhf(gr[1024 + j]);
      p.trh[e] = f2bf(sigm(gr[1536 + j]) * tanhf(c2));
      p.trc[e] = c2;
    }
    gsync();
    // ---- transition LSTM: th = LSTM(trans_emb[action], th)
    for (int tile = wv; tile < 64; tile += 32) {
      int mt = tile & 1, nt = tile >> 1;
      int m = mt * 16 + lm, n = nt * 16 + lm;
      v8f acc = bias8(p.trnB[n]);
      const bfx* rx = p.transbf + (size_t)actL[m] * 128;
      const bfx* rh = p.thh + m * 128;
      const bfx* wi = p.trnWih + (size_t)n * 128;
      const bfx* wh = p.trnWhh + (size_t)n * 128;
#pragma unroll 2
      for (int k = 0; k < 128; k += 32) acc = wmma_bf(loadA(rx, k, lh), loadB(wi + k, lh), acc);
#pragma unroll 2
      for (int k = 0; k < 128; k += 32) acc = wmma_bf(loadA(rh, k, lh), loadB(wh + k, lh), acc);
      storeC(g, acc, mt, nt, lm, lh);
    }
    __syncthreads();
    // transition LSTM elementwise (in place; all reads of th done above)
    for (int e = tid; e < 4096; e += 1024) {
      int b = e >> 7, j = e & 127;
      const float* gr = g + b * 2048;
      float c2 = sigm(gr[128 + j]) * p.thc[e] + sigm(gr[j]) * tanhf(gr[256 + j]);
      p.thh[e] = f2bf(sigm(gr[384 + j]) * tanhf(c2));
      p.thc[e] = c2;
    }
    // shift/reduce select: each thread exclusively owns element (b,j)
    for (int e = tid; e < 16384; e += 1024) {
      int b = e >> 9, j = e & 511;
      int sel = shiftL[b];
      bfx bt = p.HbH[((size_t)b * 257 + bptrL[b]) * 512 + j];
      bfx cp = p.compbf[e];
      bfx s1o = p.s1[e], s2o = p.s2[e];
      p.s1[e] = sel ? bt : cp;
      p.s2[e] = sel ? s1o : s2o;
      bfx h1o = p.sh1h[e], h2o = p.sh2h[e];
      float c1o = p.shc1[e], c2o = p.shc2[e];
      p.sh1h[e] = sel ? p.tsh[e] : p.trh[e];
      p.shc1[e] = sel ? p.tsc[e] : p.trc[e];
      p.sh2h[e] = sel ? h1o : h2o;
      p.shc2[e] = sel ? c1o : c2o;
    }
    gsync();
    if (tid < 32) bptrL[tid] = bptr2L[tid];
    __syncthreads();
  }
}

// ---------------------------------------------------------------------------
// Host launcher
// ---------------------------------------------------------------------------
extern "C" void kernel_launch(void* const* d_in, const int* in_sizes, int n_in,
                              void* d_out, int out_size, void* d_ws, size_t ws_size,
                              hipStream_t stream) {
  (void)in_sizes; (void)n_in; (void)out_size; (void)ws_size;
  const float* x        = (const float*)d_in[0];
  const float* mask     = (const float*)d_in[1];
  const float* buf_Wih  = (const float*)d_in[2];
  const float* buf_Whh  = (const float*)d_in[3];
  const float* buf_bih  = (const float*)d_in[4];
  const float* buf_bhh  = (const float*)d_in[5];
  const float* buf_begin= (const float*)d_in[6];
  const float* stk_Wih  = (const float*)d_in[7];
  const float* stk_Whh  = (const float*)d_in[8];
  const float* stk_bih  = (const float*)d_in[9];
  const float* stk_bhh  = (const float*)d_in[10];
  const float* stk_begin= (const float*)d_in[11];
  const float* trn_Wih  = (const float*)d_in[12];
  const float* trn_Whh  = (const float*)d_in[13];
  const float* trn_bih  = (const float*)d_in[14];
  const float* trn_bhh  = (const float*)d_in[15];
  const float* trn_begin= (const float*)d_in[16];
  const float* trans_emb= (const float*)d_in[17];
  const float* rel_emb  = (const float*)d_in[18];
  const float* comp_W   = (const float*)d_in[19];
  const float* comp_b   = (const float*)d_in[20];
  const float* cls_W1   = (const float*)d_in[21];
  const float* cls_b1   = (const float*)d_in[22];
  const float* cls_W2   = (const float*)d_in[23];
  const float* cls_b2   = (const float*)d_in[24];
  const float* cls_W3   = (const float*)d_in[25];
  const float* cls_b3   = (const float*)d_in[26];
  const int*   lengths  = (const int*)d_in[27];
  const int*   nsteps   = (const int*)d_in[28];
  float* out = (float*)d_out;

  char* w = (char*)d_ws;
  size_t off = 0;
  auto A = [&](size_t elems, size_t esz) -> size_t {
    size_t o = off;
    off += ((elems * esz + 255) / 256) * 256;
    return o;
  };
  size_t o_xr  = A((size_t)32 * 256 * 512, 2);
  size_t o_Hb  = A((size_t)32 * 257 * 512, 2);
  size_t o_cb  = A(16384, 4);   // phase-1 cell state
  size_t o_bWi = A(2048 * 512, 2), o_bWh = A(2048 * 512, 2);
  size_t o_sWi = A(2048 * 512, 2), o_sWh = A(2048 * 512, 2);
  size_t o_tWi = A(512 * 128, 2),  o_tWh = A(512 * 128, 2);
  size_t o_cW  = A(512 * 1280, 2);
  size_t o_W1  = A(832 * 1664, 2), o_W2 = A(832 * 832, 2), o_W3 = A(96 * 832, 2);
  size_t o_te  = A(84 * 128, 2),   o_re = A(40 * 256, 2);
  size_t o_bB  = A(2048, 4), o_sB = A(2048, 4), o_tB = A(512, 4);
  size_t o_s1  = A(16384, 2), o_s2 = A(16384, 2);
  size_t o_h1  = A(16384, 2), o_h2 = A(16384, 2);
  size_t o_c1  = A(16384, 4), o_c2 = A(16384, 4);
  size_t o_th  = A(4096, 2),  o_tc = A(4096, 4);
  size_t o_z1  = A(32 * 832, 2), o_z2 = A(32 * 832, 2);
  size_t o_cmp = A(16384, 2);
  size_t o_tsh = A(16384, 2), o_tsc = A(16384, 4);
  size_t o_trh = A(16384, 2), o_trc = A(16384, 4);

  auto BFP = [&](size_t o) { return (bfx*)(w + o); };
  auto FPo = [&](size_t o) { return (float*)(w + o); };

  auto cvt = [&](const float* s, size_t o, int n) {
    k_f2bf<<<(n + 255) / 256, 256, 0, stream>>>(s, BFP(o), n);
  };
  cvt(buf_Wih, o_bWi, 2048 * 512);
  cvt(buf_Whh, o_bWh, 2048 * 512);
  cvt(stk_Wih, o_sWi, 2048 * 512);
  cvt(stk_Whh, o_sWh, 2048 * 512);
  cvt(trn_Wih, o_tWi, 512 * 128);
  cvt(trn_Whh, o_tWh, 512 * 128);
  cvt(comp_W,  o_cW,  512 * 1280);
  cvt(cls_W1,  o_W1,  832 * 1664);
  cvt(cls_W2,  o_W2,  832 * 832);
  cvt(trans_emb, o_te, 84 * 128);
  cvt(rel_emb,   o_re, 40 * 256);
  k_padW3<<<(96 * 832 + 255) / 256, 256, 0, stream>>>(cls_W3, BFP(o_W3));
  k_biassum<<<8, 256, 0, stream>>>(buf_bih, buf_bhh, FPo(o_bB), 2048);
  k_biassum<<<8, 256, 0, stream>>>(stk_bih, stk_bhh, FPo(o_sB), 2048);
  k_biassum<<<2, 256, 0, stream>>>(trn_bih, trn_bhh, FPo(o_tB), 512);
  k_gatherx<<<(32 * 256 * 512 + 255) / 256, 256, 0, stream>>>(x, lengths, BFP(o_xr));
  k_init<<<64, 256, 0, stream>>>(stk_begin, trn_begin, BFP(o_s1), BFP(o_s2),
                                 BFP(o_h1), BFP(o_h2), FPo(o_c1), FPo(o_c2),
                                 BFP(o_th), FPo(o_tc));

  const size_t SMEM = 32 * 2048 * sizeof(float);  // 256 KB LDS gate buffer
  hipFuncSetAttribute((const void*)phase1_kernel, hipFuncAttributeMaxDynamicSharedMemorySize,
                      (int)SMEM);
  hipFuncSetAttribute((const void*)phase2_kernel, hipFuncAttributeMaxDynamicSharedMemorySize,
                      (int)SMEM);

  phase1_kernel<<<1, 1024, SMEM, stream>>>(BFP(o_xr), BFP(o_Hb), FPo(o_cb),
                                           BFP(o_bWi), BFP(o_bWh),
                                           FPo(o_bB), buf_begin, lengths);

  P2 p;
  p.HbH = BFP(o_Hb);
  p.s1 = BFP(o_s1); p.s2 = BFP(o_s2);
  p.sh1h = BFP(o_h1); p.sh2h = BFP(o_h2);
  p.thh = BFP(o_th);
  p.z1 = BFP(o_z1); p.z2 = BFP(o_z2);
  p.compbf = BFP(o_cmp);
  p.tsh = BFP(o_tsh); p.trh = BFP(o_trh);
  p.shc1 = FPo(o_c1); p.shc2 = FPo(o_c2);
  p.thc = FPo(o_tc);
  p.tsc = FPo(o_tsc); p.trc = FPo(o_trc);
  p.W1 = BFP(o_W1); p.W2 = BFP(o_W2); p.W3 = BFP(o_W3);
  p.compW = BFP(o_cW);
  p.stkWih = BFP(o_sWi); p.stkWhh = BFP(o_sWh);
  p.trnWih = BFP(o_tWi); p.trnWhh = BFP(o_tWh);
  p.transbf = BFP(o_te); p.relbf = BFP(o_re);
  p.b1 = cls_b1; p.b2 = cls_b2; p.b3 = cls_b3;
  p.compb = comp_b;
  p.stkB = FPo(o_sB); p.trnB = FPo(o_tB);
  p.mask = mask;
  p.lengths = lengths; p.nsteps = nsteps;
  p.out = out;

  phase2_kernel<<<1, 1024, SMEM, stream>>>(p);
}
